// BackwardConv2D_12987981103390
// MI455X (gfx1250) — matrix operand order repaired and checked by reading, not compiled, not run on hardware
//
#include <hip/hip_runtime.h>

// ---- problem dims ----
#define BATCH   16
#define NOUT    10
#define HDIM    64
#define WDIM    64
#define COUT    64
#define CIN     32
#define NFLAT   (HDIM*WDIM*COUT)      // 262144
#define NIN     (HDIM*WDIM*CIN)       // 131072
#define OUT_WU  ((size_t)0)
#define OUT_BU  ((size_t)BATCH*NIN*NOUT)            // 20971520
#define OUT_WL  (OUT_BU + BATCH*NOUT)               // 20971680
#define OUT_BL  (OUT_WL + (size_t)BATCH*NIN*NOUT)   // 41943200

// ---- LDS layout ----
#define KCHUNKS     18                    // 9 taps * 2 halves of 64 co, K=32 each
#define ATAB_BYTES  (KCHUNKS*2*32*16*2)   // [kc][mt][lane][16 bf16] = 36864
#define ZOFF        ATAB_BYTES            // 256B zero region
#define CELLS_OFF   (ATAB_BYTES + 256)
#define CELL_BYTES  (NOUT*COUT*2)         // [n][co] bf16 = 1280
#define NCELLS      (3*WDIM)              // 3 rows x 64 cols
// +2KB pad: lanes n>=10 intentionally read past the last cell (dead D columns)
#define SMEM_TOTAL  (CELLS_OFF + NCELLS*CELL_BYTES + 2048) // 284928

typedef __attribute__((ext_vector_type(16))) __bf16 v16bf;
typedef __attribute__((ext_vector_type(8)))  float  v8f;

__device__ __forceinline__ unsigned short f2bf(float f) {
    union { float f; unsigned u; } v; v.f = f;
    unsigned u = v.u + 0x7FFFu + ((v.u >> 16) & 1u);   // RNE
    return (unsigned short)(u >> 16);
}

__device__ __forceinline__ v16bf load_frag(const unsigned char* p) {
    union { struct { uint4 a, b; } u; v16bf v; } cv;
    cv.u.a = *(const uint4*)p;
    cv.u.b = *(const uint4*)(p + 16);
    return cv.v;
}

#define WMMA_BF16(A, B, C) __builtin_amdgcn_wmma_f32_16x16x32_bf16( \
        false, (A), false, (B), (short)0, (C), false, false)

// ---- prep: kernel (fp32 [kh][kw][ci][co]) -> bf16 A-fragments (A = K^T, M=ci, K=tap*64+co)
// A-fragment k order used here (contiguous, same as B): lane L (M = L%16 + 16*mt, g = L/16), slot j: K_local = j+16g
__global__ void prep_atab(const float* __restrict__ ker, unsigned short* __restrict__ atab) {
    int idx = blockIdx.x * 256 + threadIdx.x;        // 18432 total
    if (idx >= KCHUNKS*2*32*16) return;
    int j  = idx & 15;
    int L  = (idx >> 4) & 31;
    int mt = (idx >> 9) & 1;
    int kc = idx >> 10;
    int g  = L >> 4;
    int ci = mt*16 + (L & 15);
    int kl = j + 16*g;                               // K within 32-chunk
    int tap = kc >> 1, khalf = kc & 1;
    int co = khalf*32 + kl;
    atab[idx] = f2bf(ker[(tap*CIN + ci)*COUT + co]);
}

extern __shared__ unsigned char smem[];

__global__ __launch_bounds__(512)
void conv_bwd_wmma(const float* __restrict__ w_u, const float* __restrict__ w_l,
                   const unsigned short* __restrict__ atab_g, float* __restrict__ out) {
    const int a   = blockIdx.x;          // output row
    const int b   = blockIdx.y;          // batch
    const int set = blockIdx.z;          // 0 = upper, 1 = lower
    const float* w_in = set ? w_l : w_u;
    float* outp = out + (set ? OUT_WL : OUT_WU);

    const int tid  = threadIdx.x;
    const int lane = tid & 31;
    const int wave = __builtin_amdgcn_readfirstlane(tid >> 5);

    // stage A-fragment table
    {
        const uint* s = (const uint*)atab_g;
        uint* d = (uint*)smem;
        for (int i = tid; i < ATAB_BYTES/4; i += 512) d[i] = s[i];
    }
    if (tid < 64) ((uint*)(smem + ZOFF))[tid] = 0u;

    // stage 3 input rows (a-1..a+1), each (h,w) cell transposed to [n][co] bf16
    for (int cell = wave; cell < NCELLS; cell += 16) {
        int r = cell >> 6, col = cell & 63;
        int hp = a - 1 + r;
        if (hp < 0 || hp >= HDIM) continue;
        const float* src = w_in + ((size_t)b*NFLAT + (size_t)(hp*WDIM + col)*COUT) * NOUT;
        unsigned short* cb = (unsigned short*)(smem + CELLS_OFF + cell*CELL_BYTES);
        #pragma unroll
        for (int i = 0; i < 20; ++i) {               // 640 fp32 per cell, coalesced
            int e = i*32 + lane;
            float v = src[e];
            int co = e / 10, n = e - co*10;
            cb[n*COUT + co] = f2bf(v);
        }
    }
    __syncthreads();

    const int col0 = wave * 4;                        // 4 output columns per wave
    const int nn   = lane & 15;                       // spec index (N)
    const int g    = lane >> 4;
    const uint row_sub = (uint)nn * 128 + (uint)g * 32;  // per-lane cell offset
    const uint sub     = (uint)g * 32;
    const unsigned char* apl = smem + (uint)lane * 32;   // A-table lane base

    v8f acc[4][2];
    #pragma unroll
    for (int s = 0; s < 4; ++s) { acc[s][0] = (v8f)0.f; acc[s][1] = (v8f)0.f; }

    // K loop kept ROLLED (bounded scheduling window: no cross-iteration
    // hoisting/spills); all 6 fragment loads issued before the 8 WMMAs so
    // dscnt waits are partial and loads overlap matrix issue.
    #pragma unroll 1
    for (int kc = 0; kc < KCHUNKS; ++kc) {
        v16bf a0 = load_frag(apl + kc*2048);
        v16bf a1 = load_frag(apl + kc*2048 + 1024);
        const int tap = kc >> 1, khalf = kc & 1;
        const int kh = tap / 3, kw = tap - 3*kh;
        const int hp = a + 1 - kh;
        const bool rok = (hp >= 0) && (hp < HDIM);
        const int rloc = 2 - kh;
        // per-lane offset selected only on the (wave-uniform) halo check;
        // lanes with nn>=10 read past the cell (dead D columns, LDS padded)
        const uint basec = (uint)(CELLS_OFF + rloc*WDIM*CELL_BYTES + khalf*64) + row_sub;
        const uint zok   = (uint)(ZOFF + khalf*64) + sub;
        const int  wp0   = col0 + 1 - kw;
        const bool ok0 = rok && (wp0     >= 0) && (wp0     < WDIM);
        const bool ok1 = rok && (wp0 + 1 >= 0) && (wp0 + 1 < WDIM);
        const bool ok2 = rok && (wp0 + 2 >= 0) && (wp0 + 2 < WDIM);
        const bool ok3 = rok && (wp0 + 3 >= 0) && (wp0 + 3 < WDIM);
        const uint off0 = basec + (uint)(wp0    ) * CELL_BYTES;
        const uint off1 = basec + (uint)(wp0 + 1) * CELL_BYTES;
        const uint off2 = basec + (uint)(wp0 + 2) * CELL_BYTES;
        const uint off3 = basec + (uint)(wp0 + 3) * CELL_BYTES;
        v16bf bf0 = load_frag(smem + (ok0 ? off0 : zok));
        v16bf bf1 = load_frag(smem + (ok1 ? off1 : zok));
        v16bf bf2 = load_frag(smem + (ok2 ? off2 : zok));
        v16bf bf3 = load_frag(smem + (ok3 ? off3 : zok));
        acc[0][0] = WMMA_BF16(a0, bf0, acc[0][0]);
        acc[0][1] = WMMA_BF16(a1, bf0, acc[0][1]);
        acc[1][0] = WMMA_BF16(a0, bf1, acc[1][0]);
        acc[1][1] = WMMA_BF16(a1, bf1, acc[1][1]);
        acc[2][0] = WMMA_BF16(a0, bf2, acc[2][0]);
        acc[2][1] = WMMA_BF16(a1, bf2, acc[2][1]);
        acc[3][0] = WMMA_BF16(a0, bf3, acc[3][0]);
        acc[3][1] = WMMA_BF16(a1, bf3, acc[3][1]);
    }

    // D layout: VGPR r -> M = 8*g + r, lane%16 -> N.  M = ci within tile, N = spec.
    const size_t obase = (size_t)b * NIN;
    if (nn < NOUT) {
        #pragma unroll
        for (int s = 0; s < 4; ++s) {
            size_t sp = obase + (size_t)(a*WDIM + col0 + s) * CIN;
            #pragma unroll
            for (int mt = 0; mt < 2; ++mt)
                #pragma unroll
                for (int r = 0; r < 8; ++r) {
                    int ci = mt*16 + g*8 + r;
                    outp[(sp + ci) * NOUT + nn] = acc[s][mt][r];
                }
        }
    }
}

// bias contribution: out[b,n] = b_out[b,n] + sum_flat w[b,flat,n] * bias[flat % 64]
__global__ __launch_bounds__(256)
void bias_kernel(const float* __restrict__ w, const float* __restrict__ bvec,
                 const float* __restrict__ bias, float* __restrict__ outp) {
    const int n = blockIdx.x, b = blockIdx.y;
    const float* src = w + (size_t)b * NFLAT * NOUT + n;
    float acc = 0.f;
    for (int flat = threadIdx.x; flat < NFLAT; flat += 256)
        acc += src[(size_t)flat * NOUT] * bias[flat & 63];
    __shared__ float red[256];
    red[threadIdx.x] = acc; __syncthreads();
    for (int s = 128; s > 0; s >>= 1) {
        if (threadIdx.x < s) red[threadIdx.x] += red[threadIdx.x + s];
        __syncthreads();
    }
    if (threadIdx.x == 0) outp[b*NOUT + n] = red[0] + bvec[b*NOUT + n];
}

extern "C" void kernel_launch(void* const* d_in, const int* in_sizes, int n_in,
                              void* d_out, int out_size, void* d_ws, size_t ws_size,
                              hipStream_t stream) {
    const float* w_u  = (const float*)d_in[0];
    const float* b_u  = (const float*)d_in[1];
    const float* w_l  = (const float*)d_in[2];
    const float* b_l  = (const float*)d_in[3];
    const float* ker  = (const float*)d_in[4];
    const float* bias = (const float*)d_in[5];
    float* out = (float*)d_out;
    unsigned short* atab = (unsigned short*)d_ws;

    (void)in_sizes; (void)n_in; (void)out_size; (void)ws_size;

    hipFuncSetAttribute((const void*)conv_bwd_wmma,
                        hipFuncAttributeMaxDynamicSharedMemorySize, SMEM_TOTAL);

    prep_atab<<<72, 256, 0, stream>>>(ker, atab);
    dim3 grid(HDIM, BATCH, 2);
    conv_bwd_wmma<<<grid, 512, SMEM_TOTAL, stream>>>(w_u, w_l, atab, out);
    bias_kernel<<<dim3(NOUT, BATCH), 256, 0, stream>>>(w_u, b_u, bias, out + OUT_BU);
    bias_kernel<<<dim3(NOUT, BATCH), 256, 0, stream>>>(w_l, b_l, bias, out + OUT_BL);
}
